// GraphEmbed_35734127903526
// MI455X (gfx1250) — matrix operand order repaired and checked by reading, not compiled, not run on hardware
//
#include <hip/hip_runtime.h>

typedef __bf16 v16bf __attribute__((ext_vector_type(16)));
typedef __bf16 v8bf  __attribute__((ext_vector_type(8)));
typedef float  v8f   __attribute__((ext_vector_type(8)));

static constexpr int kAtoms  = 400000;
static constexpr int kGraphs = 10000;
static constexpr int kEdges  = 1600000;
static constexpr int kFeat   = 78;
static constexpr int kFeatP  = 96;                 // kFeat padded to a 32-multiple
static constexpr int kD      = 128;
static constexpr int kCells  = 60;
static constexpr int kBatch  = 100000;
static constexpr int kApg    = kAtoms / kGraphs;   // 40 atoms per graph (contiguous)
static constexpr int kEpg    = kEdges / kGraphs;   // 160 edges per graph (contiguous)

// ---------------------------------------------------------------------------
// WMMA fragment helpers.
// A fragment (16x32 bf16): lane L holds row (L&15); low half covers
// K = {kb..kb+7, kb+16..kb+23}, kb = kc*32 (+8 for high half) -> two 16 B runs.
// B fragment (32x16 bf16): pre-packed, 32 contiguous bytes per lane.
// ---------------------------------------------------------------------------
__device__ __forceinline__ v16bf load_frag(const __bf16* __restrict__ p)
{
    v8bf lo = *(const v8bf*)(p);
    v8bf hv = *(const v8bf*)(p + 16);
    return __builtin_shufflevector(lo, hv, 0, 1, 2, 3, 4, 5, 6, 7,
                                            8, 9, 10, 11, 12, 13, 14, 15);
}

__device__ __forceinline__ v16bf load_b_pk(const __bf16* __restrict__ Wp,
                                           long kc, int N, int n, int hi)
{
    return *(const v16bf*)&Wp[(((kc * N) + n) * 2 + hi) * 16];
}

// ---------------------------------------------------------------------------
// One-time weight pack: W (K x N, f32 row-major) -> bf16 fragments, K padded.
// out[(((kc*N + n)*2 + half)*16) + j] = W[(kc*32 + (j<8 ? half*8 + j : 16 + half*8 + j-8))*N + n]
// ---------------------------------------------------------------------------
__global__ __launch_bounds__(256) void pack_frag_kernel(
    const float* __restrict__ W, __bf16* __restrict__ out, int K, int N, long total)
{
    for (long i = blockIdx.x * 256L + threadIdx.x; i < total; i += (long)gridDim.x * 256L) {
        int  j    = (int)(i & 15);
        int  half = (int)((i >> 4) & 1);
        long rest = i >> 5;
        int  n    = (int)(rest % N);
        int  kc   = (int)(rest / N);
        int  kk   = kc * 32 + (j < 8 ? half * 8 + j : 16 + half * 8 + (j - 8));
        out[i] = (__bf16)((kk < K) ? W[(long)kk * N + n] : 0.0f);
    }
}

// ---------------------------------------------------------------------------
// Elementwise f32 -> bf16 with row padding (atom features 78 -> stride 96).
// ---------------------------------------------------------------------------
__global__ __launch_bounds__(256) void cvt_pad_kernel(
    const float* __restrict__ in, __bf16* __restrict__ out, int K, int ldo, long rows)
{
    long total = rows * ldo;
    for (long i = blockIdx.x * 256L + threadIdx.x; i < total; i += (long)gridDim.x * 256L) {
        long r = i / ldo;
        int  c = (int)(i - r * ldo);
        out[i] = (__bf16)((c < K) ? in[r * K + c] : 0.0f);
    }
}

// ---------------------------------------------------------------------------
// Per-graph SAGE mean aggregation.  Graph g owns atoms [40g,40g+40) and edges
// [160g,160g+160).  Accumulate f32 in LDS with ds_add_f32 atomics, divide by
// degree in the epilogue, store bf16 mean.  No global atomics, no memsets.
// ---------------------------------------------------------------------------
__global__ __launch_bounds__(256) void scatter_mean_kernel(
    const __bf16* __restrict__ x, int ld,
    const int* __restrict__ src, const int* __restrict__ dst,
    __bf16* __restrict__ mean)
{
    __shared__ float sagg[kApg * kD];
    __shared__ float scnt[kApg];
    const int g = blockIdx.x, tid = threadIdx.x;
    const int lane = tid & 31, wave = tid >> 5;
    const int abase = g * kApg, ebase = g * kEpg;

    for (int i = tid; i < kApg * ld; i += 256) sagg[i] = 0.0f;
    if (tid < kApg) scnt[tid] = 0.0f;
    __syncthreads();

    for (int e = wave; e < kEpg; e += 8) {
        int s = src[ebase + e] - abase;
        int d = dst[ebase + e] - abase;
        const __bf16* xs = x + (long)(abase + s) * ld;
        for (int f = lane; f < ld; f += 32)
            atomicAdd(&sagg[d * ld + f], (float)xs[f]);
        if (lane == 0) atomicAdd(&scnt[d], 1.0f);
    }
    __syncthreads();

    __bf16* mp = mean + (long)abase * ld;
    for (int i = tid; i < kApg * ld; i += 256) {
        int r = i / ld;
        float inv = 1.0f / fmaxf(scnt[r], 1.0f);
        mp[i] = (__bf16)(sagg[i] * inv);
    }
}

// ---------------------------------------------------------------------------
// N=128 GEMM, one wave per 16x128 tile, 8 wmma accumulators.  All operands
// bf16, K pre-padded (no guards).  DUAL: y = A1@W1 + A2@W2 + bias (SAGE).
// mode: 0 = bias only, 1 = relu(LN(y)), 2 = LN(relu(y)).
// ---------------------------------------------------------------------------
template <bool DUAL>
__global__ __launch_bounds__(256) void gemm_ln_kernel(
    const __bf16* __restrict__ A1, int lda1, const __bf16* __restrict__ W1p,
    const __bf16* __restrict__ A2, int lda2, const __bf16* __restrict__ W2p,
    const float* __restrict__ bias,
    const float* __restrict__ lng, const float* __restrict__ lnb,
    __bf16* __restrict__ out, int M, int kch, int mode)
{
    const int tid = threadIdx.x, lane = tid & 31, wave = tid >> 5;
    const int hi = lane >> 4, rr = lane & 15;
    const int rowbase = (blockIdx.x * 8 + wave) * 16;
    if (rowbase >= M) return;                    // M % 16 == 0 at all call sites
    const int rA = rowbase + rr;
    const __bf16* rowA1 = A1 + (long)rA * lda1;
    const __bf16* rowA2 = DUAL ? (A2 + (long)rA * lda2) : nullptr;

    v8f acc[8] = {};
    for (int kc = 0; kc < kch; ++kc) {
        if (kc + 1 < kch) {
            __builtin_prefetch(&W1p[((long)(kc + 1) * kD) * 32], 0, 1);
            if (DUAL) __builtin_prefetch(&W2p[((long)(kc + 1) * kD) * 32], 0, 1);
        }
        const int kb = kc * 32 + (hi ? 8 : 0);
        v16bf a1 = load_frag(rowA1 + kb);
        v16bf a2;
        if (DUAL) a2 = load_frag(rowA2 + kb);
#pragma unroll
        for (int t = 0; t < 8; ++t) {
            int n = t * 16 + rr;
            v16bf b1 = load_b_pk(W1p, kc, kD, n, hi);
            acc[t] = __builtin_amdgcn_wmma_f32_16x16x32_bf16(false, a1, false, b1,
                                                             (short)0, acc[t], false, false);
            if (DUAL) {
                v16bf b2 = load_b_pk(W2p, kc, kD, n, hi);
                acc[t] = __builtin_amdgcn_wmma_f32_16x16x32_bf16(false, a2, false, b2,
                                                                 (short)0, acc[t], false, false);
            }
        }
    }

    // -------- epilogue --------
    float bn[8], gl[8], bl[8];
#pragma unroll
    for (int t = 0; t < 8; ++t) {
        int n = t * 16 + rr;
        bn[t] = bias ? bias[n] : 0.0f;
        gl[t] = (mode && lng) ? lng[n] : 1.0f;
        bl[t] = (mode && lnb) ? lnb[n] : 0.0f;
    }
#pragma unroll
    for (int t = 0; t < 8; ++t)
#pragma unroll
        for (int r = 0; r < 8; ++r) {
            float x = acc[t][r] + bn[t];
            if (mode == 2) x = fmaxf(x, 0.0f);
            acc[t][r] = x;
        }
    if (mode) {
        // row (r + hi*8) lives in one 16-lane half: reduce with xor 1,2,4,8
#pragma unroll
        for (int r = 0; r < 8; ++r) {
            float s = 0.0f, s2 = 0.0f;
#pragma unroll
            for (int t = 0; t < 8; ++t) { float x = acc[t][r]; s += x; s2 += x * x; }
#pragma unroll
            for (int m = 1; m <= 8; m <<= 1) { s += __shfl_xor(s, m); s2 += __shfl_xor(s2, m); }
            float mean = s * (1.0f / 128.0f);
            float var  = s2 * (1.0f / 128.0f) - mean * mean;
            float rs   = rsqrtf(var + 1e-5f);
#pragma unroll
            for (int t = 0; t < 8; ++t) {
                float x = (acc[t][r] - mean) * rs * gl[t] + bl[t];
                if (mode == 1) x = fmaxf(x, 0.0f);
                acc[t][r] = x;
            }
        }
    }
#pragma unroll
    for (int t = 0; t < 8; ++t) {
        int n = t * 16 + rr;
#pragma unroll
        for (int r = 0; r < 8; ++r)
            out[(long)(rowbase + r + hi * 8) * kD + n] = (__bf16)acc[t][r];
    }
}

// ---------------------------------------------------------------------------
__global__ void pool_max_kernel(const __bf16* __restrict__ x, __bf16* __restrict__ pooled)
{
    int g = blockIdx.x, f = threadIdx.x;
    const __bf16* xp = x + (long)g * kApg * kD + f;
    float m = -3.4028235e38f;
    for (int a = 0; a < kApg; ++a) m = fmaxf(m, (float)xp[a * kD]);
    pooled[(long)g * kD + f] = (__bf16)m;
}

// ---------------------------------------------------------------------------
__global__ void cell_ln_kernel(const float* __restrict__ g, const float* __restrict__ b,
                               __bf16* __restrict__ cl)
{
    __shared__ float s1[128], s2[128];
    int row = blockIdx.x, f = threadIdx.x;
    float v = (f == row) ? 1.0f : 0.0f;
    s1[f] = v; s2[f] = v * v;
    __syncthreads();
    for (int off = 64; off > 0; off >>= 1) {
        if (f < off) { s1[f] += s1[f + off]; s2[f] += s2[f + off]; }
        __syncthreads();
    }
    float mean = s1[0] / 128.0f;
    float var  = s2[0] / 128.0f - mean * mean;
    cl[row * 128 + f] = (__bf16)((v - mean) * rsqrtf(var + 1e-5f) * g[f] + b[f]);
}

// ---------------------------------------------------------------------------
// Fused trunk, 32 rows per block (two 16-row tiles per wave: each B fragment
// feeds TWO wmma ops).  All phases alias one 64 KB LDS pool (gather 24K ->
// h1 64K -> h2 32K), separated by barriers.  Gather is pure 16 B copies.
// ---------------------------------------------------------------------------
__global__ __launch_bounds__(256) void trunk_kernel(
    const __bf16* __restrict__ drug, const __bf16* __restrict__ cl,
    const int* __restrict__ indices,
    const __bf16* __restrict__ t0Wp, const float* __restrict__ t0b,
    const float* __restrict__ g0, const float* __restrict__ b0,
    const __bf16* __restrict__ t1Wp, const float* __restrict__ t1b,
    const float* __restrict__ pW, const float* __restrict__ pb,
    float* __restrict__ out)
{
    __shared__ __align__(16) char smpool[65536];
    __bf16* sm_t  = (__bf16*)smpool;   // phase 1: 32 x 384  (24 KB)
    __bf16* sm_h1 = (__bf16*)smpool;   // phase 2: 32 x 1024 (64 KB)
    __bf16* sm_h2 = (__bf16*)smpool;   // phase 3: 32 x 512  (32 KB)
    const int tid = threadIdx.x, lane = tid & 31, wave = tid >> 5;
    const int hi = lane >> 4, rr = lane & 15;
    const int rowbase = blockIdx.x * 32;

    // gather: 32 rows x 48 chunks of 8 bf16 (16 B vector copies, no cvt)
    for (int i = tid; i < 32 * 48; i += 256) {
        int r = i / 48, cc = (i - r * 48) * 8;
        const int* ix = indices + (long)(rowbase + r) * 3;
        const __bf16* sp;
        if (cc < 128)      sp = drug + (long)ix[0] * 128 + cc;
        else if (cc < 256) sp = drug + (long)ix[1] * 128 + (cc - 128);
        else               sp = cl + (long)ix[2] * 128 + (cc - 256);
        *(uint4*)&sm_t[r * 384 + cc] = *(const uint4*)sp;
    }
    __syncthreads();

    // GEMM1: 32x384 @ 384x1024; wave covers 128 cols x 2 row-tiles
    v8f acc[2][8] = {};
    for (int kc = 0; kc < 12; ++kc) {
        if (kc + 1 < 12)
            __builtin_prefetch(&t0Wp[(((long)(kc + 1) * 1024) + wave * 128) * 32], 0, 1);
        v16bf a0 = load_frag(&sm_t[(rr)      * 384 + kc * 32 + (hi ? 8 : 0)]);
        v16bf a1 = load_frag(&sm_t[(16 + rr) * 384 + kc * 32 + (hi ? 8 : 0)]);
#pragma unroll
        for (int t = 0; t < 8; ++t) {
            int n = wave * 128 + t * 16 + rr;
            v16bf b = load_b_pk(t0Wp, kc, 1024, n, hi);
            acc[0][t] = __builtin_amdgcn_wmma_f32_16x16x32_bf16(false, a0, false, b,
                                                                (short)0, acc[0][t], false, false);
            acc[1][t] = __builtin_amdgcn_wmma_f32_16x16x32_bf16(false, a1, false, b,
                                                                (short)0, acc[1][t], false, false);
        }
    }
    __syncthreads();   // done reading sm_t; pool becomes h1
#pragma unroll
    for (int m = 0; m < 2; ++m)
#pragma unroll
        for (int t = 0; t < 8; ++t) {
            int n = wave * 128 + t * 16 + rr;
            float bb = t0b[n];
#pragma unroll
            for (int r = 0; r < 8; ++r)
                sm_h1[(m * 16 + r + hi * 8) * 1024 + n] = (__bf16)fmaxf(acc[m][t][r] + bb, 0.0f);
        }
    __syncthreads();

    // LayerNorm over 1024 per row (four rows per wave)
#pragma unroll
    for (int q = 0; q < 4; ++q) {
        int row = wave * 4 + q;
        float s = 0.0f, s2 = 0.0f;
        for (int n = lane; n < 1024; n += 32) {
            float x = (float)sm_h1[row * 1024 + n]; s += x; s2 += x * x;
        }
#pragma unroll
        for (int m = 1; m <= 16; m <<= 1) { s += __shfl_xor(s, m); s2 += __shfl_xor(s2, m); }
        float mean = s * (1.0f / 1024.0f);
        float var  = s2 * (1.0f / 1024.0f) - mean * mean;
        float rs   = rsqrtf(var + 1e-5f);
        for (int n = lane; n < 1024; n += 32) {
            float x = (float)sm_h1[row * 1024 + n];
            sm_h1[row * 1024 + n] = (__bf16)((x - mean) * rs * g0[n] + b0[n]);
        }
    }
    __syncthreads();

    // GEMM2: 32x1024 @ 1024x512; wave covers 64 cols x 2 row-tiles
    v8f acc2[2][4] = {};
    for (int kc = 0; kc < 32; ++kc) {
        if (kc + 1 < 32)
            __builtin_prefetch(&t1Wp[(((long)(kc + 1) * 512) + wave * 64) * 32], 0, 1);
        v16bf a0 = load_frag(&sm_h1[(rr)      * 1024 + kc * 32 + (hi ? 8 : 0)]);
        v16bf a1 = load_frag(&sm_h1[(16 + rr) * 1024 + kc * 32 + (hi ? 8 : 0)]);
#pragma unroll
        for (int t = 0; t < 4; ++t) {
            int n = wave * 64 + t * 16 + rr;
            v16bf b = load_b_pk(t1Wp, kc, 512, n, hi);
            acc2[0][t] = __builtin_amdgcn_wmma_f32_16x16x32_bf16(false, a0, false, b,
                                                                 (short)0, acc2[0][t], false, false);
            acc2[1][t] = __builtin_amdgcn_wmma_f32_16x16x32_bf16(false, a1, false, b,
                                                                 (short)0, acc2[1][t], false, false);
        }
    }
    __syncthreads();   // done reading sm_h1; pool becomes h2
#pragma unroll
    for (int m = 0; m < 2; ++m)
#pragma unroll
        for (int t = 0; t < 4; ++t) {
            int n = wave * 64 + t * 16 + rr;
            float bb = t1b[n];
#pragma unroll
            for (int r = 0; r < 8; ++r)
                sm_h2[(m * 16 + r + hi * 8) * 512 + n] = (__bf16)fmaxf(acc2[m][t][r] + bb, 0.0f);
        }
    __syncthreads();

    // pred: out[row] = h2[row] . pred_W + pred_b   (four rows per wave)
#pragma unroll
    for (int q = 0; q < 4; ++q) {
        int row = wave * 4 + q;
        float s = 0.0f;
        for (int n = lane; n < 512; n += 32) s += (float)sm_h2[row * 512 + n] * pW[n];
#pragma unroll
        for (int m = 1; m <= 16; m <<= 1) s += __shfl_xor(s, m);
        if (lane == 0) out[rowbase + row] = s + pb[0];
    }
}

// ---------------------------------------------------------------------------
extern "C" void kernel_launch(void* const* d_in, const int* in_sizes, int n_in,
                              void* d_out, int out_size, void* d_ws, size_t ws_size,
                              hipStream_t stream)
{
    const float* atom = (const float*)d_in[0];
    const int* eidx = (const int*)d_in[1];
    const int* src = eidx;
    const int* dst = eidx + kEdges;
    const int* indices = (const int*)d_in[3];     // d_in[2] = batch (contiguous, unused)
    const float* W0l = (const float*)d_in[4];
    const float* b0  = (const float*)d_in[5];
    const float* W0r = (const float*)d_in[6];
    const float* W1l = (const float*)d_in[7];
    const float* b1  = (const float*)d_in[8];
    const float* W1r = (const float*)d_in[9];
    const float* W2l = (const float*)d_in[10];
    const float* b2  = (const float*)d_in[11];
    const float* W2r = (const float*)d_in[12];
    const float* ln0g = (const float*)d_in[13];
    const float* ln0b = (const float*)d_in[14];
    const float* ln1g = (const float*)d_in[15];
    const float* ln1b = (const float*)d_in[16];
    const float* headW = (const float*)d_in[17];
    const float* headb = (const float*)d_in[18];
    const float* hlng = (const float*)d_in[19];
    const float* hlnb = (const float*)d_in[20];
    const float* clng = (const float*)d_in[21];
    const float* clnb = (const float*)d_in[22];
    const float* t0W = (const float*)d_in[23];
    const float* t0b = (const float*)d_in[24];
    const float* tg0 = (const float*)d_in[25];
    const float* tb0 = (const float*)d_in[26];
    const float* t1W = (const float*)d_in[27];
    const float* t1b = (const float*)d_in[28];
    const float* pW  = (const float*)d_in[29];
    const float* pb  = (const float*)d_in[30];

    char* ws = (char*)d_ws;
    __bf16* mean_bf = (__bf16*)(ws);                  // 102,400,000 B (stride 96 or 128)
    __bf16* xA      = (__bf16*)(ws + 102400000L);     // 102,400,000 B
    __bf16* xB      = (__bf16*)(ws + 204800000L);     // 102,400,000 B
    __bf16* atom_bf = (__bf16*)(ws + 307200000L);     //  76,800,000 B (stride 96)
    __bf16* pooled  = (__bf16*)(ws + 384000000L);     //   2,560,000 B
    __bf16* drug    = (__bf16*)(ws + 386560000L);     //   2,560,000 B
    __bf16* cl      = (__bf16*)(ws + 389120000L);     //      15,360 B
    // packed bf16 weight fragments
    __bf16* p_t0  = (__bf16*)(ws + 389135360L);       // 786,432 B
    __bf16* p_t1  = (__bf16*)(ws + 389921792L);       // 1,048,576 B
    __bf16* p_W0l = (__bf16*)(ws + 390970368L);       // 24,576 B
    __bf16* p_W0r = (__bf16*)(ws + 390994944L);
    __bf16* p_W1l = (__bf16*)(ws + 391019520L);       // 32,768 B
    __bf16* p_W1r = (__bf16*)(ws + 391052288L);
    __bf16* p_W2l = (__bf16*)(ws + 391085056L);
    __bf16* p_W2r = (__bf16*)(ws + 391117824L);
    __bf16* p_hW  = (__bf16*)(ws + 391150592L);

    dim3 blk(256);

    // ---- pack weights into WMMA B-fragment order (bf16, K zero-padded) ----
    auto pack = [&](const float* W, __bf16* out, int K, int N) {
        long total = (long)((K + 31) / 32) * N * 32;
        int blocks = (int)((total + 255) / 256);
        if (blocks > 4096) blocks = 4096;
        pack_frag_kernel<<<blocks, blk, 0, stream>>>(W, out, K, N, total);
    };
    pack(W0l, p_W0l, kFeat, kD);  pack(W0r, p_W0r, kFeat, kD);
    pack(W1l, p_W1l, kD, kD);     pack(W1r, p_W1r, kD, kD);
    pack(W2l, p_W2l, kD, kD);     pack(W2r, p_W2r, kD, kD);
    pack(headW, p_hW, kD, kD);
    pack(t0W, p_t0, 384, 1024);
    pack(t1W, p_t1, 1024, 512);

    // ---- atom features f32 -> bf16, padded stride 96 ----
    cvt_pad_kernel<<<4096, blk, 0, stream>>>(atom, atom_bf, kFeat, kFeatP, kAtoms);

    // ---- SAGE layer 0 (78 -> 128), epilogue relu(LN(.)) ----
    scatter_mean_kernel<<<kGraphs, blk, 0, stream>>>(atom_bf, kFeatP, src, dst, mean_bf);
    gemm_ln_kernel<true><<<kAtoms / 128, blk, 0, stream>>>(
        mean_bf, kFeatP, p_W0l, atom_bf, kFeatP, p_W0r, b0, ln0g, ln0b, xA, kAtoms, 3, 1);
    // ---- SAGE layer 1 (128 -> 128) ----
    scatter_mean_kernel<<<kGraphs, blk, 0, stream>>>(xA, kD, src, dst, mean_bf);
    gemm_ln_kernel<true><<<kAtoms / 128, blk, 0, stream>>>(
        mean_bf, kD, p_W1l, xA, kD, p_W1r, b1, ln1g, ln1b, xB, kAtoms, 4, 1);
    // ---- SAGE layer 2 (128 -> 128), no epilogue ----
    scatter_mean_kernel<<<kGraphs, blk, 0, stream>>>(xB, kD, src, dst, mean_bf);
    gemm_ln_kernel<true><<<kAtoms / 128, blk, 0, stream>>>(
        mean_bf, kD, p_W2l, xB, kD, p_W2r, b2, nullptr, nullptr, xA, kAtoms, 4, 0);
    // ---- graph max-pool ----
    pool_max_kernel<<<kGraphs, kD, 0, stream>>>(xA, pooled);
    // ---- drug head: LN(relu(pooled @ headW + headb)) ----
    gemm_ln_kernel<false><<<(kGraphs + 127) / 128, blk, 0, stream>>>(
        pooled, kD, p_hW, nullptr, 0, nullptr, headb, hlng, hlnb, drug, kGraphs, 4, 2);
    // ---- cell-line table ----
    cell_ln_kernel<<<kCells, kD, 0, stream>>>(clng, clnb, cl);
    // ---- fused trunk MLP ----
    trunk_kernel<<<kBatch / 32, blk, 0, stream>>>(drug, cl, indices,
                                                  p_t0, t0b, tg0, tb0,
                                                  p_t1, t1b, pW, pb, (float*)d_out);
}
